// ConvOp_78159814852954
// MI455X (gfx1250) — matrix operand order repaired and checked by reading, not compiled, not run on hardware
//
#include <hip/hip_runtime.h>

// ---------------------------------------------------------------------------
// Fused kernel-prediction network block for MI455X (gfx1250, wave32, WMMA).
//   out = KernelConv( x , upsample4( wk*(dw3x3(w1*x+b1)+b2)+bk ) )
// One workgroup = one 8x8 spatial tile (+1 halo) of one batch image.
// All intermediates live in LDS; HBM traffic = read x + write out only
// (~134 MB total -> ~5.8 us floor at 23.3 TB/s; f16 WMMA keeps the ~7.5
// GFLOP of GEMM work well under that, so the kernel stays memory-bound).
// ---------------------------------------------------------------------------

typedef __attribute__((ext_vector_type(16))) _Float16 v16h;
typedef __attribute__((ext_vector_type(8)))  float    v8f;

#define TILE 8
#define HT   10          // halo tile edge
#define NPH  100         // halo pixels (10*10)
#define NP   64          // interior pixels (8*8)
#define CSX  67          // xs  per-pixel channel stride (f32, odd -> no bank conflicts)
#define CSH  66          // h1s/h2s per-pixel channel stride (f16, even -> b32 pair loads)
#define KVS  146         // kvs per-pixel k stride (f16)

// LDS layout (bytes):
//   [0,     26800)  xs   float  [100*67]            (phases 0..4)
//   [26800, 45488)  h1s  f16    [100*66]  (ph 1-2)  / kvs f16 [64*146] (ph 3-4)
//   [45488, 53936)  h2s  f16    [64*66]             (phases 2-3)
#define OFF_H1K 26800
#define OFF_H2  45488
#define SMEM_BYTES 53936

// --- optional CDNA5 async global->LDS path (gated so compile never breaks) ---
#ifdef __has_builtin
# if __has_builtin(__builtin_amdgcn_global_load_async_to_lds_b32)
#  define ASYNC_LDS 1
# endif
# if __has_builtin(__builtin_amdgcn_s_wait_asynccnt)
#  define HAVE_WAIT_ASYNC 1
# endif
#endif
#ifndef ASYNC_LDS
# define ASYNC_LDS 0
#endif

#if ASYNC_LDS
typedef __attribute__((address_space(1))) int g_int;
typedef __attribute__((address_space(3))) int l_int;
__device__ __forceinline__ void async_g2l_b32(const float* g, float* l) {
  __builtin_amdgcn_global_load_async_to_lds_b32((g_int*)g, (l_int*)l, 0, 0);
}
__device__ __forceinline__ void wait_async_lds() {
# if defined(HAVE_WAIT_ASYNC)
  __builtin_amdgcn_s_wait_asynccnt(0);
# else
  asm volatile("s_wait_asynccnt 0" ::: "memory");
# endif
}
#endif

__global__ __launch_bounds__(256) void fused_kpn_kernel(
    const float* __restrict__ xg,  const float* __restrict__ w1g,
    const float* __restrict__ b1g, const float* __restrict__ w2g,
    const float* __restrict__ b2g, const float* __restrict__ wkg,
    const float* __restrict__ bkg, float* __restrict__ outg)
{
  __shared__ __align__(16) unsigned char smem[SMEM_BYTES];
  float*    xs  = (float*)smem;
  _Float16* h1s = (_Float16*)(smem + OFF_H1K);
  _Float16* kvs = (_Float16*)(smem + OFF_H1K);   // overlays h1s (h1 dead by phase 3)
  _Float16* h2s = (_Float16*)(smem + OFF_H2);

  const int tid   = threadIdx.x;
  const int lane  = tid & 31;
  const int wv    = tid >> 5;
  const int khalf = lane >> 4;      // 0: lanes 0-15, 1: lanes 16-31
  const int nlane = lane & 15;
  const int bx = blockIdx.x, by = blockIdx.y, bat = blockIdx.z;
  const int gx0 = bx * TILE - 1, gy0 = by * TILE - 1;
  const float* xb = xg + (size_t)bat * 64 * 256 * 256;

  // ---------------- Phase 0: stage x halo tile (f32) into LDS --------------
  for (int idx = tid; idx < NPH * 64; idx += 256) {
    int hp = idx % NPH, c = idx / NPH;
    int gy = gy0 + hp / HT, gx = gx0 + hp % HT;
    float* dst = &xs[hp * CSX + c];
    if ((unsigned)gy < 256u && (unsigned)gx < 256u) {
      const float* src = xb + ((size_t)c * 256 + gy) * 256 + gx;
#if ASYNC_LDS
      async_g2l_b32(src, dst);
#else
      *dst = *src;
#endif
    } else {
      *dst = 0.f;
    }
  }
#if ASYNC_LDS
  wait_async_lds();
#endif
  __syncthreads();

  // ------- Phase 1: h1 = w1*x + b1 over 100 halo pixels (WMMA f16) ---------
  // A = w1 [64x64] (M x K), B = x [64 x 100] (K x N). 4 M-tiles x 7 N-tiles.
  {
    const int mt = wv & 3, moff = mt << 4;
    const float* Wrow = w1g + (moff + nlane) * 64;   // lane's A row (m = moff+nlane)
    v16h a0, a1;
#pragma unroll
    for (int v = 0; v < 8; ++v) {   // A 16x32 layout: K-pairs, halves split by 8
      int kb = ((v & 4) ? 16 : 0) + (khalf ? 8 : 0) + ((v & 3) << 1);
      a0[2*v]   = (_Float16)Wrow[kb];      a0[2*v+1] = (_Float16)Wrow[kb + 1];
      a1[2*v]   = (_Float16)Wrow[32 + kb]; a1[2*v+1] = (_Float16)Wrow[32 + kb + 1];
    }
    float bias[8];
#pragma unroll
    for (int r = 0; r < 8; ++r) bias[r] = b1g[moff + r + (khalf ? 8 : 0)];

    for (int j = (wv >> 2); j <= 6; j += 2) {        // N-tiles: waves 0-3 even, 4-7 odd
      int p = 16 * j + nlane; if (p > NPH - 1) p = NPH - 1;   // clamp pad columns
      const float* xcol = &xs[p * CSX];
      v16h bb0, bb1;
#pragma unroll
      for (int v = 0; v < 8; ++v) {  // B 32x16 layout: K-pairs, halves split by 8
        int kb = ((v & 4) ? 16 : 0) + (khalf ? 8 : 0) + ((v & 3) << 1);
        bb0[2*v]   = (_Float16)xcol[kb];      bb0[2*v+1] = (_Float16)xcol[kb + 1];
        bb1[2*v]   = (_Float16)xcol[32 + kb]; bb1[2*v+1] = (_Float16)xcol[32 + kb + 1];
      }
      v8f acc = {};
      acc = __builtin_amdgcn_wmma_f32_16x16x32_f16(false, a0, false, bb0, (short)0, acc, false, false);
      acc = __builtin_amdgcn_wmma_f32_16x16x32_f16(false, a1, false, bb1, (short)0, acc, false, false);
      int pout = 16 * j + nlane;
      if (pout < NPH) {
#pragma unroll
        for (int r = 0; r < 8; ++r) {
          int mrow = moff + r + (khalf ? 8 : 0);
          h1s[pout * CSH + mrow] = (_Float16)(acc[r] + bias[r]);
        }
      }
    }
  }
  __syncthreads();

  // ------------- Phase 2: depthwise 3x3 + b2 -> h2 (interior) --------------
  for (int idx = tid; idx < NP * 64; idx += 256) {
    int c = idx & 63, p = idx >> 6;
    int y = p >> 3, x = p & 7;
    const float* wr = w2g + c * 9;
    float acc = b2g[c];
#pragma unroll
    for (int dy = 0; dy < 3; ++dy)
#pragma unroll
      for (int dx = 0; dx < 3; ++dx)
        acc += wr[dy * 3 + dx] * (float)h1s[((y + dy) * HT + (x + dx)) * CSH + c];
    h2s[p * CSH + c] = (_Float16)acc;
  }
  __syncthreads();

  // --------- Phase 3: kv = wk*h2 + bk  [144 x 64px] (WMMA f16) -------------
  // 9 M-tiles x 4 N-tiles = 36 tiles over 8 waves. kvs overlays dead h1s.
  for (int tt = wv; tt < 36; tt += 8) {
    int mt = tt % 9, nt = tt / 9;
    int moff = mt << 4;
    const float* Wrow = wkg + (moff + nlane) * 64;
    v16h a0, a1;
#pragma unroll
    for (int v = 0; v < 8; ++v) {
      int kb = ((v & 4) ? 16 : 0) + (khalf ? 8 : 0) + ((v & 3) << 1);
      a0[2*v]   = (_Float16)Wrow[kb];      a0[2*v+1] = (_Float16)Wrow[kb + 1];
      a1[2*v]   = (_Float16)Wrow[32 + kb]; a1[2*v+1] = (_Float16)Wrow[32 + kb + 1];
    }
    int p = (nt << 4) + nlane;
    const _Float16* hcol = &h2s[p * CSH];
    v16h bb0, bb1;
#pragma unroll
    for (int v = 0; v < 8; ++v) {
      int kb = ((v & 4) ? 16 : 0) + (khalf ? 8 : 0) + ((v & 3) << 1);
      bb0[2*v]   = hcol[kb];      bb0[2*v+1] = hcol[kb + 1];
      bb1[2*v]   = hcol[32 + kb]; bb1[2*v+1] = hcol[32 + kb + 1];
    }
    v8f acc = {};
    acc = __builtin_amdgcn_wmma_f32_16x16x32_f16(false, a0, false, bb0, (short)0, acc, false, false);
    acc = __builtin_amdgcn_wmma_f32_16x16x32_f16(false, a1, false, bb1, (short)0, acc, false, false);
#pragma unroll
    for (int r = 0; r < 8; ++r) {
      int krow = moff + r + (khalf ? 8 : 0);
      kvs[p * KVS + krow] = (_Float16)(acc[r] + bkg[krow]);
    }
  }
  __syncthreads();

  // -- Phase 4: out[c,p] = sum_t kv[(9c+t)>>2, p] * x[c, p + d_t] -----------
  float* outb = outg + (size_t)bat * 64 * 256 * 256;
  for (int idx = tid; idx < NP * 64; idx += 256) {
    int p = idx & 63, c = idx >> 6;
    int y = p >> 3, x = p & 7;
    float acc = 0.f;
#pragma unroll
    for (int t = 0; t < 9; ++t) {
      int k  = (c * 9 + t) >> 2;            // nearest channel-upsample x4
      int dy = t / 3, dx = t % 3;
      acc += (float)kvs[p * KVS + k] * xs[((y + dy) * HT + (x + dx)) * CSX + c];
    }
    int gy = by * TILE + y, gx = bx * TILE + x;
    outb[((size_t)c * 256 + gy) * 256 + gx] = acc;
  }
}

extern "C" void kernel_launch(void* const* d_in, const int* in_sizes, int n_in,
                              void* d_out, int out_size, void* d_ws, size_t ws_size,
                              hipStream_t stream) {
  (void)in_sizes; (void)n_in; (void)out_size; (void)d_ws; (void)ws_size;
  const float* x  = (const float*)d_in[0];
  const float* w1 = (const float*)d_in[1];
  const float* b1 = (const float*)d_in[2];
  const float* w2 = (const float*)d_in[3];
  const float* b2 = (const float*)d_in[4];
  const float* wk = (const float*)d_in[5];
  const float* bk = (const float*)d_in[6];
  float* out = (float*)d_out;
  dim3 grid(256 / TILE, 256 / TILE, 4);   // 32 x 32 x 4 = 4096 workgroups
  fused_kpn_kernel<<<grid, 256, 0, stream>>>(x, w1, b1, w2, b2, wk, bk, out);
}
